// ConvNeXtParallelMoELoRA_10299331575852
// MI455X (gfx1250) — compile-verified, matrix operand order checked by reading, not run on hardware
//
#include <hip/hip_runtime.h>
#include <hip/hip_bf16.h>

// ---------------------------------------------------------------------------
// Types for CDNA5 WMMA (wave32)
// ---------------------------------------------------------------------------
typedef __bf16 bf16_t;
typedef __attribute__((ext_vector_type(16))) __bf16 v16bf;
typedef __attribute__((ext_vector_type(8)))  float  v8f;
typedef __attribute__((ext_vector_type(4)))  unsigned int u32x4;
typedef __attribute__((ext_vector_type(8)))  int i32x8;
typedef __attribute__((ext_vector_type(4)))  int i32x4;

union FragBF {
    v16bf v;
    u32x4 q[2];
};

static __device__ __forceinline__ bf16_t f2bf(float f) {
    // round-to-nearest-even float -> bf16
    unsigned u = __builtin_bit_cast(unsigned, f);
    unsigned r = (u + 0x7FFFu + ((u >> 16) & 1u)) >> 16;
    unsigned short s = (unsigned short)r;
    return __builtin_bit_cast(bf16_t, s);
}

static __device__ __forceinline__ float gelu_exact(float v) {
    return 0.5f * v * (1.0f + erff(v * 0.70710678118654752440f));
}

// ---------------------------------------------------------------------------
// TDM: 2D tile load Global -> LDS (bf16 elements).
//   tile = tile_rows x 64 elements; LDS dest padded +4 dwords per 32-dword
//   row  => LDS row stride 144 bytes (bank-conflict-free b128 frag reads).
//   Descriptor per CDNA5 ISA; 6-arg builtin
//   (uint32x4 g0, int32x8 g1, int32x4 g2, int32x4 g3, int32x8 g4, int cpol).
// ---------------------------------------------------------------------------
static __device__ __forceinline__ void tdm_load_tile(
    unsigned lds_addr, const bf16_t* gptr, int k_len, int rows_total,
    int tile_rows, int row_stride_elems) {
    unsigned long long ga = (unsigned long long)(size_t)gptr;

    u32x4 g0;
    g0[0] = 1u;                                    // count=1, user desc
    g0[1] = lds_addr;                              // LDS byte address
    g0[2] = (unsigned)(ga & 0xFFFFFFFFull);        // global_addr[31:0]
    g0[3] = (unsigned)((ga >> 32) & 0x1FFFFFFull)  // global_addr[56:32]
            | (2u << 30);                          // type = 2 ("image")

    i32x8 g1;
    g1[0] = (1 << 16)      // data_size = 1 -> 2 bytes
          | (1 << 20)      // pad_enable
          | (4 << 22)      // pad_interval = 4 -> every 32 dwords (one row)
          | (3 << 25);     // pad_amount  = 3 -> 4 dwords (16 bytes)
    g1[1] = (k_len & 0xFFFF) << 16;                          // tensor_dim0 lo
    g1[2] = ((k_len >> 16) & 0xFFFF) | ((rows_total & 0xFFFF) << 16);
    g1[3] = ((rows_total >> 16) & 0xFFFF) | (64 << 16);      // tile_dim0 = 64
    g1[4] = (tile_rows & 0xFFFF);                            // tile_dim1
    g1[5] = row_stride_elems;                                // dim0 stride lo
    g1[6] = 0;
    g1[7] = 0;

    i32x4 gz4 = {0, 0, 0, 0};
    i32x8 gz8 = {0, 0, 0, 0, 0, 0, 0, 0};
    __builtin_amdgcn_tensor_load_to_lds(g0, g1, gz4, gz4, gz8, 0);
}

// ---------------------------------------------------------------------------
// Conversion kernels
// ---------------------------------------------------------------------------
__global__ __launch_bounds__(256) void cvt_f32_bf16_kernel(
    const float* __restrict__ in, bf16_t* __restrict__ out, int n) {
    int i = (blockIdx.x * 256 + threadIdx.x) * 4;
    if (i + 3 < n) {
        out[i + 0] = f2bf(in[i + 0]);
        out[i + 1] = f2bf(in[i + 1]);
        out[i + 2] = f2bf(in[i + 2]);
        out[i + 3] = f2bf(in[i + 3]);
    }
}

// in: [R x C] row-major f32  ->  out: [C x R] row-major bf16
__global__ __launch_bounds__(256) void transpose_cvt_kernel(
    const float* __restrict__ in, bf16_t* __restrict__ out, int R, int C) {
    int idx = blockIdx.x * 256 + threadIdx.x;
    if (idx < R * C) {
        int c = idx / R;
        int r = idx - c * R;
        out[idx] = f2bf(in[(size_t)r * C + c]);
    }
}

// ---------------------------------------------------------------------------
// bf16 WMMA GEMM with TDM double-buffered LDS staging:
//   C[M,N] = act(A[M,K] @ B^T[N,K] + bias[N])
//   A row-major (lda = K stride), B stored N-major (ldb = K stride).
//   Block: 256 threads = 8 waves; wave tile 32x64 (2x4 frags, 64 accum
//   VGPRs); block tile 256x64 (waves stacked along M, B frags broadcast
//   from LDS). K slab = 64: TDM stages A(256x64) + B(64x64) bf16 tiles with
//   144B padded row stride; wave 0 issues TDM + s_wait_tensorcnt, barrier
//   releases the block. Next slab's TDM overlaps current slab's WMMAs.
//   Fragment traffic: 12 ds_load_b128 per 8 WMMAs per wave.
// ---------------------------------------------------------------------------
#define LDS_ROW   144               // 128B data + 16B pad
#define A_TILE_B  (256 * LDS_ROW)   // 36864
#define B_TILE_B  (64  * LDS_ROW)   // 9216
#define BUF_B     (A_TILE_B + B_TILE_B)

template <bool GELU_OUT>
__global__ __launch_bounds__(256) void gemm_wmma_bf16_kernel(
    const bf16_t* __restrict__ A, const bf16_t* __restrict__ B,
    const float* __restrict__ bias, void* __restrict__ Cout,
    int M, int N, int K, int lda, int ldb, int ldc) {
    __shared__ __align__(16) unsigned char smem[2 * BUF_B];   // 90 KB

    const int wave = threadIdx.x >> 5;   // 0..7 -> M direction
    const int lane = threadIdx.x & 31;
    const int half = lane >> 4;          // K-half select per ISA layout
    const int r    = lane & 15;          // row within 16

    const int mBlk = blockIdx.y * 256;
    const int nBlk = blockIdx.x * 64;

    const unsigned ldsBase = (unsigned)(size_t)&smem[0];

    v8f acc[2][4] = {};

    // Prologue: stage first K slab
    if (threadIdx.x < 32) {
        tdm_load_tile(ldsBase,            A + (size_t)mBlk * lda, K, M, 256, lda);
        tdm_load_tile(ldsBase + A_TILE_B, B + (size_t)nBlk * ldb, K, N, 64,  ldb);
    }

    int buf = 0;
    for (int k0 = 0; k0 < K; k0 += 64) {
        if (threadIdx.x < 32) __builtin_amdgcn_s_wait_tensorcnt(0);
        __syncthreads();   // current slab resident; prev compute finished

        if (k0 + 64 < K && threadIdx.x < 32) {
            const unsigned nb = ldsBase + (buf ^ 1) * BUF_B;
            tdm_load_tile(nb,            A + (size_t)mBlk * lda + (k0 + 64), K, M, 256, lda);
            tdm_load_tile(nb + A_TILE_B, B + (size_t)nBlk * ldb + (k0 + 64), K, N, 64,  ldb);
        }

        const unsigned char* aBase = &smem[buf * BUF_B];
        const unsigned char* bBase = aBase + A_TILE_B;
        const int aRow0 = wave * 32 + r;      // block-relative A row
        const int bRow0 = r;                  // block-relative B row

#pragma unroll
        for (int s = 0; s < 2; ++s) {         // two 32-K WMMA steps per slab
            const int ko = s * 64 + half * 16;  // byte offset within row
            FragBF a[2], b[4];
#pragma unroll
            for (int i = 0; i < 2; ++i) {
                const size_t ao = (size_t)(aRow0 + i * 16) * LDS_ROW + ko;
                a[i].q[0] = *(const u32x4*)(aBase + ao);
                a[i].q[1] = *(const u32x4*)(aBase + ao + 32);
            }
#pragma unroll
            for (int j = 0; j < 4; ++j) {
                const size_t bo = (size_t)(bRow0 + j * 16) * LDS_ROW + ko;
                b[j].q[0] = *(const u32x4*)(bBase + bo);
                b[j].q[1] = *(const u32x4*)(bBase + bo + 32);
            }
#pragma unroll
            for (int i = 0; i < 2; ++i)
#pragma unroll
                for (int j = 0; j < 4; ++j)
                    acc[i][j] = __builtin_amdgcn_wmma_f32_16x16x32_bf16(
                        false, a[i].v, false, b[j].v, (short)0, acc[i][j],
                        false, false);
        }
        buf ^= 1;
    }

    // Store per documented f32 16x16 C/D layout:
    //   lane L: col = L&15 ; VGPR v -> row v (lanes 0-15) / row v+8 (16-31)
    const int rowOff = half * 8;
#pragma unroll
    for (int i = 0; i < 2; ++i) {
#pragma unroll
        for (int j = 0; j < 4; ++j) {
            const int m0 = mBlk + wave * 32 + i * 16 + rowOff;
            const int n  = nBlk + j * 16 + r;
            const float bv = bias[n];
#pragma unroll
            for (int v = 0; v < 8; ++v) {
                float val = acc[i][j][v] + bv;
                if (GELU_OUT) {
                    val = gelu_exact(val);
                    ((bf16_t*)Cout)[(size_t)(m0 + v) * ldc + n] = f2bf(val);
                } else {
                    ((float*)Cout)[(size_t)(m0 + v) * ldc + n] = val;
                }
            }
        }
    }
}

// ---------------------------------------------------------------------------
// Router + rank-8 LoRA MoE: one wave32 per token, out[t,:] += moe_out[t,:]
// ---------------------------------------------------------------------------
__global__ __launch_bounds__(256) void moe_lora_kernel(
    const float* __restrict__ x, const float* __restrict__ router_w,
    const float* __restrict__ router_b, const float* __restrict__ w_down,
    const float* __restrict__ w_up, float* __restrict__ out, int T) {
    constexpr int D = 768, E = 3, R = 8;
    const int lane = threadIdx.x & 31;
    const int t = blockIdx.x * 8 + (threadIdx.x >> 5);
    if (t >= T) return;

    const float* xt = x + (size_t)t * D;

    float racc[E] = {0.f, 0.f, 0.f};
    float dacc[E * R];
#pragma unroll
    for (int i = 0; i < E * R; ++i) dacc[i] = 0.f;

    for (int j = lane; j < D; j += 32) {
        const float xv = xt[j];
#pragma unroll
        for (int e = 0; e < E; ++e) racc[e] += xv * router_w[j * E + e];
#pragma unroll
        for (int e = 0; e < E; ++e) {
            const float* wd = w_down + ((size_t)e * D + j) * R;
#pragma unroll
            for (int rr = 0; rr < R; ++rr) dacc[e * R + rr] += xv * wd[rr];
        }
    }

    // wave32 xor-reduction: every lane ends with the full sum
#pragma unroll
    for (int e = 0; e < E; ++e) {
        float v = racc[e];
        for (int o = 16; o > 0; o >>= 1) v += __shfl_xor(v, o, 32);
        racc[e] = v + router_b[e];
    }
#pragma unroll
    for (int i = 0; i < E * R; ++i) {
        float v = dacc[i];
        for (int o = 16; o > 0; o >>= 1) v += __shfl_xor(v, o, 32);
        dacc[i] = v;
    }

    // softmax over 3 logits
    float mx = fmaxf(racc[0], fmaxf(racc[1], racc[2]));
    float e0 = expf(racc[0] - mx), e1 = expf(racc[1] - mx), e2 = expf(racc[2] - mx);
    float inv_se = 1.0f / (e0 + e1 + e2);
    float p[E] = {e0 * inv_se, e1 * inv_se, e2 * inv_se};

    // top-2 of 3 == drop argmin, renormalize survivors
    int amin = 0;
    if (p[1] < p[amin]) amin = 1;
    if (p[2] < p[amin]) amin = 2;
    float comb[E], psum = 0.f;
#pragma unroll
    for (int e = 0; e < E; ++e) { comb[e] = (e == amin) ? 0.f : p[e]; psum += comb[e]; }
    const float invp = 1.0f / (psum + 1e-6f);
#pragma unroll
    for (int e = 0; e < E; ++e) comb[e] *= invp;

    float act[E * R];
#pragma unroll
    for (int i = 0; i < E * R; ++i) act[i] = gelu_exact(dacc[i]);

    // SCALING = ALPHA / R = 1.0
    for (int d = lane; d < D; d += 32) {
        float m = 0.f;
#pragma unroll
        for (int e = 0; e < E; ++e) {
            float s = 0.f;
#pragma unroll
            for (int rr = 0; rr < R; ++rr)
                s += act[e * R + rr] * w_up[((size_t)e * R + rr) * D + d];
            m += comb[e] * s;
        }
        out[(size_t)t * D + d] += m;
    }
}

// ---------------------------------------------------------------------------
// Launch
// ---------------------------------------------------------------------------
extern "C" void kernel_launch(void* const* d_in, const int* in_sizes, int n_in,
                              void* d_out, int out_size, void* d_ws, size_t ws_size,
                              hipStream_t stream) {
    constexpr int D = 768, H = 3072;
    const float* x        = (const float*)d_in[0];
    const float* w1       = (const float*)d_in[1];
    const float* b1       = (const float*)d_in[2];
    const float* w2       = (const float*)d_in[3];
    const float* b2       = (const float*)d_in[4];
    const float* router_w = (const float*)d_in[5];
    const float* router_b = (const float*)d_in[6];
    const float* w_down   = (const float*)d_in[7];
    const float* w_up     = (const float*)d_in[8];
    float* out = (float*)d_out;

    const int T = in_sizes[0] / D;  // 12544 = 49 * 256

    // Workspace carve-up (256B aligned)
    char* ws = (char*)d_ws;
    size_t off = 0;
    auto take = [&](size_t bytes) -> void* {
        void* p = ws + off;
        off += (bytes + 255) & ~(size_t)255;
        return p;
    };
    bf16_t* xbf = (bf16_t*)take((size_t)T * D * sizeof(bf16_t));
    bf16_t* w1T = (bf16_t*)take((size_t)H * D * sizeof(bf16_t));  // [H][D]
    bf16_t* w2T = (bf16_t*)take((size_t)D * H * sizeof(bf16_t));  // [D][H]
    bf16_t* hbf = (bf16_t*)take((size_t)T * H * sizeof(bf16_t));  // [T][H]

    // 1) precision conversion + weight transposition (once per launch)
    cvt_f32_bf16_kernel<<<(T * D) / 1024, 256, 0, stream>>>(x, xbf, T * D);
    transpose_cvt_kernel<<<(D * H + 255) / 256, 256, 0, stream>>>(w1, w1T, D, H);
    transpose_cvt_kernel<<<(D * H + 255) / 256, 256, 0, stream>>>(w2, w2T, H, D);

    // 2) h = gelu(x @ w1 + b1)   [T x H], bf16
    gemm_wmma_bf16_kernel<true><<<dim3(H / 64, T / 256), 256, 0, stream>>>(
        xbf, w1T, b1, hbf, T, H, /*K=*/D, /*lda=*/D, /*ldb=*/D, /*ldc=*/H);

    // 3) out = h @ w2 + b2       [T x D], f32
    gemm_wmma_bf16_kernel<false><<<dim3(D / 64, T / 256), 256, 0, stream>>>(
        hbf, w2T, b2, out, T, D, /*K=*/H, /*lda=*/H, /*ldb=*/H, /*ldc=*/D);

    // 4) out += MoE LoRA path
    moe_lora_kernel<<<T / 8, 256, 0, stream>>>(
        x, router_w, router_b, w_down, w_up, out, T);
}
